// MultiheadSelfAttention_49606872269052
// MI455X (gfx1250) — matrix operand order repaired and checked by reading, not compiled, not run on hardware
//
#include <hip/hip_runtime.h>
#include <cstdint>

// ---------------------------------------------------------------------------
// MI455X (gfx1250) multi-head self-attention, bf16 WMMA path + TDM staging.
// Shapes fixed by the reference: B=2, S=2048, D=1024, H=16, d_k=64.
// ---------------------------------------------------------------------------

typedef __attribute__((ext_vector_type(16))) __bf16        v16bf;
typedef __attribute__((ext_vector_type(8)))  float         v8f;
typedef __attribute__((ext_vector_type(4)))  unsigned int  v4u;
typedef __attribute__((ext_vector_type(8)))  int           v8i;
typedef __attribute__((ext_vector_type(4)))  int           v4i;

#define D_MODEL 1024
#define SEQ     2048
#define BATCH   2
#define NHEAD   16
#define DK      64

// Tensor Data Mover availability (this toolchain: 6-arg clang-23 form).
#if defined(__has_builtin)
#  if __has_builtin(__builtin_amdgcn_tensor_load_to_lds) && \
      __has_builtin(__builtin_amdgcn_s_wait_tensorcnt)
#    define USE_TDM 1
#  endif
#endif
#ifndef USE_TDM
#  define USE_TDM 0
#endif

__device__ __forceinline__ __bf16 f2bf(float f) {
  union { float f; uint32_t u; } in; in.f = f;
  uint32_t u = in.u;
  uint32_t r = (u + 0x7FFFu + ((u >> 16) & 1u)) >> 16;   // round-to-nearest-even
  union { unsigned short s; __bf16 b; } out; out.s = (unsigned short)r;
  return out.b;
}

// ---------------------------------------------------------------------------
// fp32 -> bf16 conversion (vectorized by 4; all sizes are multiples of 4)
// ---------------------------------------------------------------------------
__global__ __launch_bounds__(256) void convert_f32_to_bf16(
    const float* __restrict__ src, __bf16* __restrict__ dst, int n4) {
  int i = blockIdx.x * blockDim.x + threadIdx.x;
  if (i < n4) {
    float4 v = ((const float4*)src)[i];
    __bf16 o[4] = { f2bf(v.x), f2bf(v.y), f2bf(v.z), f2bf(v.w) };
    ((uint2*)dst)[i] = *(uint2*)o;
  }
}

__device__ __forceinline__ void storeC(float* p, float v)  { *p = v; }
__device__ __forceinline__ void storeC(__bf16* p, float v) { *p = f2bf(v); }

#if USE_TDM
// ---------------------------------------------------------------------------
// TDM: DMA a 128x32 bf16 tile from a row-major [rows x K] matrix into LDS,
// padding each 64B tile row with 16B so LDS rows land on a 40-element pitch
// (bank-conflict-free fragment reads). Issued by one wave; TENSORcnt tracked.
// D# layout per CDNA5 ISA.
// ---------------------------------------------------------------------------
__device__ __forceinline__ void tdm_load_tile_128x32(
    const __bf16* gsrc, int K, int M, unsigned ldsByteAddr) {
  unsigned long long ga = (unsigned long long)(uintptr_t)gsrc;
  v4u g0 = {};
  g0[0] = 1u;                                        // count=1, user mode
  g0[1] = ldsByteAddr;                               // lds_addr (bytes)
  g0[2] = (unsigned)(ga & 0xFFFFFFFFu);              // global_addr[31:0]
  g0[3] = (unsigned)((ga >> 32) & 0x01FFFFFFu)       // global_addr[56:32]
          | (2u << 30);                              // type=2 ("image")
  v8i g1 = {};
  g1[0] = (int)((1u << 16)       // data_size = 1 -> 2 bytes
              | (1u << 20)       // pad_enable
              | (3u << 22)       // pad_interval: 16 DWORDs (one 64B tile row)
              | (3u << 25));     // pad_amount: 4 DWORDs (16B)
  g1[1] = (int)(((unsigned)K & 0xFFFFu) << 16);      // tensor_dim0[15:0]
  g1[2] = (int)((((unsigned)K >> 16) & 0xFFFFu)      // tensor_dim0[31:16]
              | (((unsigned)M & 0xFFFFu) << 16));    // tensor_dim1[15:0]
  g1[3] = (int)((((unsigned)M >> 16) & 0xFFFFu)      // tensor_dim1[31:16]
              | (32u << 16));                        // tile_dim0 = 32
  g1[4] = (int)128u;                                 // tile_dim1 = 128
  g1[5] = (int)(unsigned)K;                          // tensor_dim0_stride[31:0]
  g1[6] = 0;                                         // stride[47:32] | dim1_stride lo
  g1[7] = 0;
  v4i gz4 = {};
  v8i gz8 = {};
  // 6-arg form: (g0, g1, group2, group3, <extra group>, cpol)
  __builtin_amdgcn_tensor_load_to_lds(g0, g1, gz4, gz4, gz8, 0);
}
#endif

// ---------------------------------------------------------------------------
// C[M,N] = A[M,K] * Bw[N,K]^T   (bf16 in, f32 accumulate, OutT out)
// Block tile 128x128, 8 waves, each wave = 64x32 (4x2 WMMA 16x16 tiles).
// K tiles double-buffered in LDS; staged by the Tensor Data Mover when
// available, otherwise by cooperative b128 loads.
// Requires M%128==0, N%128==0, K%32==0 (true for all uses here).
// ---------------------------------------------------------------------------
template <typename OutT>
__global__ __launch_bounds__(256) void gemm_nt_wmma(
    const __bf16* __restrict__ A,
    const __bf16* __restrict__ Bw,
    OutT* __restrict__ C,
    int M, int N, int K)
{
  __shared__ __align__(16) __bf16 As[2][128][40];
  __shared__ __align__(16) __bf16 Bs[2][128][40];

  const int tid   = threadIdx.x;
  const int lane  = tid & 31;
  const int wave  = tid >> 5;
  const int l16   = lane & 15;
  const int lh    = lane >> 4;      // lane half: 0 or 1
  const int waveM = wave >> 2;      // 0..1  (64-row slab)
  const int waveN = wave & 3;       // 0..3  (32-col slab)

  const int mBase = blockIdx.y * 128;
  const int nBase = blockIdx.x * 128;

  v8f zero = {};
  v8f acc[4][2];
#pragma unroll
  for (int i = 0; i < 4; ++i)
#pragma unroll
    for (int j = 0; j < 2; ++j) acc[i][j] = zero;

  const int NT = K / 32;

#if USE_TDM
  if (wave == 0) {   // uniform per wave; TDM ignores EXEC
    tdm_load_tile_128x32(A  + (size_t)mBase * K, K, M,
                         (unsigned)(uintptr_t)&As[0][0][0]);
    tdm_load_tile_128x32(Bw + (size_t)nBase * K, K, N,
                         (unsigned)(uintptr_t)&Bs[0][0][0]);
  }
#endif

  for (int kt = 0; kt < NT; ++kt) {
    const int buf = kt & 1;
#if USE_TDM
    if (wave == 0) {
      if (kt + 1 < NT) {   // prefetch next K-tile into the other buffer
        const int k1 = (kt + 1) * 32;
        tdm_load_tile_128x32(A  + (size_t)mBase * K + k1, K, M,
                             (unsigned)(uintptr_t)&As[buf ^ 1][0][0]);
        tdm_load_tile_128x32(Bw + (size_t)nBase * K + k1, K, N,
                             (unsigned)(uintptr_t)&Bs[buf ^ 1][0][0]);
        __builtin_amdgcn_s_wait_tensorcnt(2);   // current tile's 2 ops done
      } else {
        __builtin_amdgcn_s_wait_tensorcnt(0);
      }
    }
    __syncthreads();                            // publish tile kt
#else
    const int k0 = kt * 32;
    __syncthreads();
#pragma unroll
    for (int c = tid; c < 512; c += 256) {
      int row = c >> 2;
      int kc  = (c & 3) * 8;
      *(uint4*)&As[buf][row][kc] = *(const uint4*)&A [(size_t)(mBase + row) * K + k0 + kc];
      *(uint4*)&Bs[buf][row][kc] = *(const uint4*)&Bw[(size_t)(nBase + row) * K + k0 + kc];
    }
    __syncthreads();
#endif

    // A fragment (16x32 bf16): lane half lh covers K = lh*8..+7 and +16.
    v16bf aF[4], bF[2];
    const int khalf = lh * 8;
#pragma unroll
    for (int mi = 0; mi < 4; ++mi) {
      const __bf16* p = &As[buf][waveM * 64 + mi * 16 + l16][0];
      *(uint4*)&aF[mi]       = *(const uint4*)&p[khalf];
      *((uint4*)&aF[mi] + 1) = *(const uint4*)&p[khalf + 16];
    }
    // B fragment (32x16): lane half lh covers K = lh*8..+7 and +16.
    const int kh = lh * 8;
#pragma unroll
    for (int ni = 0; ni < 2; ++ni) {
      const __bf16* p = &Bs[buf][waveN * 32 + ni * 16 + l16][0];
      *(uint4*)&bF[ni]       = *(const uint4*)&p[kh];
      *((uint4*)&bF[ni] + 1) = *(const uint4*)&p[kh + 16];
    }
#pragma unroll
    for (int mi = 0; mi < 4; ++mi)
#pragma unroll
      for (int ni = 0; ni < 2; ++ni)
        acc[mi][ni] = __builtin_amdgcn_wmma_f32_16x16x32_bf16(
            false, aF[mi], false, bF[ni], (short)0, acc[mi][ni], false, false);

#if USE_TDM
    __syncthreads();   // all waves done reading buf before TDM overwrites it
#endif
  }

  // Epilogue: C VGPR r holds row r (lanes 0-15) / r+8 (lanes 16-31), col=l16.
#pragma unroll
  for (int mi = 0; mi < 4; ++mi) {
    const int rowBase = mBase + waveM * 64 + mi * 16 + lh * 8;
#pragma unroll
    for (int ni = 0; ni < 2; ++ni) {
      const int col = nBase + waveN * 32 + ni * 16 + l16;
#pragma unroll
      for (int r = 0; r < 8; ++r)
        storeC(&C[(size_t)(rowBase + r) * N + col], acc[mi][ni][r]);
    }
  }
}

// ---------------------------------------------------------------------------
// Flash attention: one block = one (b,h) and 128 query rows; 8 waves, each
// owning 16 query rows with a 16x64 f32 context accumulator. Keys iterate in
// tiles of 32 with online softmax. Causal mask + 0.125 scale + the reference's
// +1.0 bias on unmasked entries.
// ---------------------------------------------------------------------------
__global__ __launch_bounds__(256) void attn_flash(
    const __bf16* __restrict__ Q,
    const __bf16* __restrict__ Km,
    const __bf16* __restrict__ V,
    __bf16* __restrict__ Ctx)
{
  __shared__ __align__(16) __bf16 Vs[DK][40];     // V tile transposed: [dk][key]
  __shared__ __align__(16) __bf16 Ps[8][16][40];  // per-wave P tile (16x32)

  const int b     = blockIdx.z;
  const int h     = blockIdx.y;
  const int qBase = blockIdx.x * 128;
  const int tid   = threadIdx.x;
  const int lane  = tid & 31;
  const int wave  = tid >> 5;
  const int l16   = lane & 15;
  const int lh    = lane >> 4;

  const size_t base = ((size_t)b * SEQ) * D_MODEL + (size_t)h * DK;
  const __bf16* Qg = Q   + base;   // row stride D_MODEL
  const __bf16* Kg = Km  + base;
  const __bf16* Vg = V   + base;
  __bf16*       Cg = Ctx + base;

  // Q fragments: 16 rows x 64 dk -> two 16x32 A-fragments, direct from global.
  v16bf qa[2];
  {
    const int qRow = qBase + wave * 16 + l16;
    const __bf16* p = Qg + (size_t)qRow * D_MODEL;
    const int khalf = lh * 8;
#pragma unroll
    for (int kc = 0; kc < 2; ++kc) {
      *(uint4*)&qa[kc]       = *(const uint4*)&p[kc * 32 + khalf];
      *((uint4*)&qa[kc] + 1) = *(const uint4*)&p[kc * 32 + khalf + 16];
    }
  }

  v8f zero = {};
  v8f ctxAcc[4];
#pragma unroll
  for (int i = 0; i < 4; ++i) ctxAcc[i] = zero;
  float mrow[8], lrow[8];
#pragma unroll
  for (int r = 0; r < 8; ++r) { mrow[r] = -3.0e38f; lrow[r] = 0.0f; }

  const float scale = 0.125f;              // 1/sqrt(64)
  const int   nKT   = qBase / 32 + 4;      // cover keys 0 .. qBase+127

  for (int kt = 0; kt < nKT; ++kt) {
    const int keyBase = kt * 32;
    __syncthreads();
    { // Stage V tile transposed: 32 keys x 64 dk -> Vs[dk][key].
      const int key = tid >> 3;
      const int c0  = (tid & 7) * 8;
      __bf16 tmp[8];
      *(uint4*)tmp = *(const uint4*)&Vg[(size_t)(keyBase + key) * D_MODEL + c0];
#pragma unroll
      for (int j = 0; j < 8; ++j) Vs[c0 + j][key] = tmp[j];
    }
    __syncthreads();

    // Scores: S = Q (16x64) * K^T (64x32) via 2 n-tiles x 2 k-chunks.
    v8f sc[2]; sc[0] = zero; sc[1] = zero;
#pragma unroll
    for (int ni = 0; ni < 2; ++ni) {
      const int rowK = keyBase + ni * 16 + l16;
      const __bf16* kp = Kg + (size_t)rowK * D_MODEL + lh * 8;
#pragma unroll
      for (int kc = 0; kc < 2; ++kc) {
        v16bf bk;   // B fragment: dk lh*8..+7 and +16 of one K row
        *(uint4*)&bk       = *(const uint4*)&kp[kc * 32];
        *((uint4*)&bk + 1) = *(const uint4*)&kp[kc * 32 + 16];
        sc[ni] = __builtin_amdgcn_wmma_f32_16x16x32_bf16(
            false, qa[kc], false, bk, (short)0, sc[ni], false, false);
      }
    }

    // Online softmax per query row (row r+8*lh lives in the 16-lane half).
#pragma unroll
    for (int r = 0; r < 8; ++r) {
      const int qrow = qBase + wave * 16 + r + lh * 8;
      float s0 = sc[0][r] * scale + 1.0f;
      float s1 = sc[1][r] * scale + 1.0f;
      if (keyBase + l16 > qrow)      s0 = -3.0e38f;
      if (keyBase + 16 + l16 > qrow) s1 = -3.0e38f;
      float mx = fmaxf(s0, s1);
#pragma unroll
      for (int off = 8; off >= 1; off >>= 1)
        mx = fmaxf(mx, __shfl_xor(mx, off, 32));
      const float mnew  = fmaxf(mrow[r], mx);
      const float alpha = __expf(mrow[r] - mnew);
      const float p0    = __expf(s0 - mnew);
      const float p1    = __expf(s1 - mnew);
      float rs = p0 + p1;
#pragma unroll
      for (int off = 8; off >= 1; off >>= 1)
        rs += __shfl_xor(rs, off, 32);
      lrow[r] = lrow[r] * alpha + rs;
      mrow[r] = mnew;
#pragma unroll
      for (int ni = 0; ni < 4; ++ni) ctxAcc[ni][r] *= alpha;
      sc[0][r] = p0; sc[1][r] = p1;
    }

    // Reshape P from C-layout to A-fragment layout via per-wave LDS tile.
#pragma unroll
    for (int r = 0; r < 8; ++r) {
      Ps[wave][r + lh * 8][l16]      = f2bf(sc[0][r]);
      Ps[wave][r + lh * 8][16 + l16] = f2bf(sc[1][r]);
    }
    __builtin_amdgcn_wave_barrier();   // LDS is in-order per wave
    v16bf pa;
    {
      const __bf16* p = &Ps[wave][l16][0];
      const int khalf = lh * 8;
      *(uint4*)&pa       = *(const uint4*)&p[khalf];
      *((uint4*)&pa + 1) = *(const uint4*)&p[khalf + 16];
    }

    // ctx += P (16x32) * V (32x64): B fragments from transposed Vs.
#pragma unroll
    for (int ni = 0; ni < 4; ++ni) {
      v16bf bv;
      const __bf16* p = &Vs[ni * 16 + l16][lh * 8];
      *(uint4*)&bv       = *(const uint4*)&p[0];
      *((uint4*)&bv + 1) = *(const uint4*)&p[16];
      ctxAcc[ni] = __builtin_amdgcn_wmma_f32_16x16x32_bf16(
          false, pa, false, bv, (short)0, ctxAcc[ni], false, false);
    }
  }

  // Normalize and store context (bf16) into [B,S,D] at head slice.
#pragma unroll
  for (int r = 0; r < 8; ++r) {
    const int   row = qBase + wave * 16 + r + lh * 8;
    const float inv = 1.0f / lrow[r];
    __bf16* cp = Cg + (size_t)row * D_MODEL;
#pragma unroll
    for (int ni = 0; ni < 4; ++ni)
      cp[ni * 16 + l16] = f2bf(ctxAcc[ni][r] * inv);
  }
}

// ---------------------------------------------------------------------------
// Launch: convert -> QKV GEMMs -> flash attention -> output GEMM (f32 out).
// ---------------------------------------------------------------------------
extern "C" void kernel_launch(void* const* d_in, const int* in_sizes, int n_in,
                              void* d_out, int out_size, void* d_ws, size_t ws_size,
                              hipStream_t stream)
{
  const float* x  = (const float*)d_in[0];
  const float* qw = (const float*)d_in[1];
  const float* kw = (const float*)d_in[2];
  const float* vw = (const float*)d_in[3];
  const float* ow = (const float*)d_in[4];
  float* out = (float*)d_out;
  (void)in_sizes; (void)n_in; (void)out_size; (void)ws_size;

  const size_t nx = (size_t)BATCH * SEQ * D_MODEL;   // 4 Mi elems
  const size_t nw = (size_t)D_MODEL * D_MODEL;       // 1 Mi elems

  char* ws = (char*)d_ws;
  auto carve = [&](size_t bytes) {
    char* p = ws; ws += (bytes + 255) & ~(size_t)255; return p;
  };
  __bf16* xb  = (__bf16*)carve(nx * 2);
  __bf16* wqb = (__bf16*)carve(nw * 2);
  __bf16* wkb = (__bf16*)carve(nw * 2);
  __bf16* wvb = (__bf16*)carve(nw * 2);
  __bf16* wob = (__bf16*)carve(nw * 2);
  __bf16* Qb  = (__bf16*)carve(nx * 2);
  __bf16* Kb  = (__bf16*)carve(nx * 2);
  __bf16* Vb  = (__bf16*)carve(nx * 2);
  __bf16* Cb  = (__bf16*)carve(nx * 2);

  {
    const int n4 = (int)(nx / 4);
    const int w4 = (int)(nw / 4);
    convert_f32_to_bf16<<<(n4 + 255) / 256, 256, 0, stream>>>(x,  xb,  n4);
    convert_f32_to_bf16<<<(w4 + 255) / 256, 256, 0, stream>>>(qw, wqb, w4);
    convert_f32_to_bf16<<<(w4 + 255) / 256, 256, 0, stream>>>(kw, wkb, w4);
    convert_f32_to_bf16<<<(w4 + 255) / 256, 256, 0, stream>>>(vw, wvb, w4);
    convert_f32_to_bf16<<<(w4 + 255) / 256, 256, 0, stream>>>(ow, wob, w4);
  }

  const int M = BATCH * SEQ;                 // 4096
  dim3 gg(D_MODEL / 128, M / 128);           // (8, 32)

  gemm_nt_wmma<__bf16><<<gg, 256, 0, stream>>>(xb, wqb, Qb, M, D_MODEL, D_MODEL);
  gemm_nt_wmma<__bf16><<<gg, 256, 0, stream>>>(xb, wkb, Kb, M, D_MODEL, D_MODEL);
  gemm_nt_wmma<__bf16><<<gg, 256, 0, stream>>>(xb, wvb, Vb, M, D_MODEL, D_MODEL);

  attn_flash<<<dim3(SEQ / 128, NHEAD, BATCH), 256, 0, stream>>>(Qb, Kb, Vb, Cb);

  gemm_nt_wmma<float><<<gg, 256, 0, stream>>>(Cb, wob, out, M, D_MODEL, D_MODEL);
}
